// MultiQueryAttention_42167988912975
// MI455X (gfx1250) — matrix operand order repaired and checked by reading, not compiled, not run on hardware
//
#include <hip/hip_runtime.h>
#include <hip/hip_bf16.h>

// MQA reference collapses to: out = (x @ Wv + bv) @ Wo_sum + bo
// (scores are constant over the broadcast head axis -> softmax uniform -> attn out == v)

typedef __attribute__((ext_vector_type(16))) __bf16 bf16x16;
typedef __attribute__((ext_vector_type(8)))  float  f32x8;

#define DMODEL 1024
#define HDIM   64
#define NHEAD  16
#define NROWS  16384      // B*S
#define ROWS_PER_WG 32
#define LDS_PITCH 68      // padded f32 pitch for the 64-wide v tile

// ---------------------------------------------------------------------------
// Prep: WvT[n][k] = Wv[k][n]                (bf16, 64x1024)
//       WoT[n][d] = sum_h Wo[h*64+d][n]     (bf16, 1024x64)
// ---------------------------------------------------------------------------
__global__ void prep_weights(const float* __restrict__ Wv,
                             const float* __restrict__ Wo,
                             __bf16* __restrict__ WvT,
                             __bf16* __restrict__ WoT)
{
    int tid = blockIdx.x * blockDim.x + threadIdx.x;
    if (tid < HDIM * DMODEL) {
        int n = tid >> 10;            // 0..63
        int k = tid & 1023;           // 0..1023
        WvT[tid] = (__bf16)Wv[k * HDIM + n];
    } else if (tid < 2 * HDIM * DMODEL) {
        int t = tid - HDIM * DMODEL;
        int n = t >> 6;               // 0..1023
        int d = t & 63;               // 0..63
        float s = 0.f;
#pragma unroll
        for (int h2 = 0; h2 < NHEAD; ++h2)
            s += Wo[(h2 * HDIM + d) * DMODEL + n];
        WoT[t] = (__bf16)s;
    }
}

// ---------------------------------------------------------------------------
// Fused, K-split + N-split for occupancy: 512 WGs x 8 wave32 = 4096 waves.
// Stage 1: wave (rt, ks) accumulates v-partial[16x64] over K in [ks*256, ks*256+256)
//          -> f32 partials to LDS.
// Stage 2: wave (rt, cb) sums 4 partials (+bv), builds bf16 A-frags in regs,
//          computes out[16 x 256] = v @ Wo_sum (+bo).
// ---------------------------------------------------------------------------
__global__ __launch_bounds__(256) void mqa_fused(
    const float*  __restrict__ x,
    const __bf16* __restrict__ WvT,   // [64][1024]
    const float*  __restrict__ bv,
    const __bf16* __restrict__ WoT,   // [1024][64]
    const float*  __restrict__ bo,
    float*        __restrict__ out)
{
    __shared__ float vp[4][ROWS_PER_WG][LDS_PITCH];   // ~34.8 KB

    const int lane = threadIdx.x & 31;
    const int wave = threadIdx.x >> 5;    // 0..7
    const int h    = lane >> 4;           // lane half
    const int l16  = lane & 15;

    const long blockrow = (long)blockIdx.x * ROWS_PER_WG;

    // ------------------ stage 1: partial v = x @ Wv over one K slice ------------------
    const int rt1   = wave >> 2;          // row stripe 0/1
    const int slice = wave & 3;           // K slice 0..3

    f32x8 acc[4] = {};
    const float* xrow = x + (blockrow + rt1 * 16 + l16) * DMODEL;

    for (int kt = slice * 8; kt < slice * 8 + 8; ++kt) {
        const int k0 = kt * 32 + 16 * h;
        const float4* p0 = (const float4*)(xrow + k0);
        const float4* p1 = (const float4*)(xrow + k0 + 8);
        float4 f0 = p0[0], f1 = p0[1];
        float4 f2 = p1[0], f3 = p1[1];
        __builtin_prefetch(xrow + kt * 32 + 32, 0, 3);

        bf16x16 afrag;
        afrag[0]  = (__bf16)f0.x; afrag[1]  = (__bf16)f0.y;
        afrag[2]  = (__bf16)f0.z; afrag[3]  = (__bf16)f0.w;
        afrag[4]  = (__bf16)f1.x; afrag[5]  = (__bf16)f1.y;
        afrag[6]  = (__bf16)f1.z; afrag[7]  = (__bf16)f1.w;
        afrag[8]  = (__bf16)f2.x; afrag[9]  = (__bf16)f2.y;
        afrag[10] = (__bf16)f2.z; afrag[11] = (__bf16)f2.w;
        afrag[12] = (__bf16)f3.x; afrag[13] = (__bf16)f3.y;
        afrag[14] = (__bf16)f3.z; afrag[15] = (__bf16)f3.w;

        // load all 4 B fragments first so they clause together and overlap WMMA
        bf16x16 bfrag[4];
#pragma unroll
        for (int t = 0; t < 4; ++t)
            bfrag[t] = *(const bf16x16*)(WvT + (t * 16 + l16) * DMODEL + kt * 32 + 16 * h);
#pragma unroll
        for (int t = 0; t < 4; ++t)
            acc[t] = __builtin_amdgcn_wmma_f32_16x16x32_bf16(
                false, afrag, false, bfrag[t], (short)0, acc[t], false, false);
    }

    // partial v (f32, row-major) -> LDS
#pragma unroll
    for (int t = 0; t < 4; ++t)
#pragma unroll
        for (int r = 0; r < 8; ++r)
            vp[slice][rt1 * 16 + 8 * h + r][t * 16 + l16] = acc[t][r];
    __syncthreads();

    // ------------------ stage 2: out = v @ Wo_sum over one 256-col block ------------------
    const int rt2 = wave >> 2;            // row stripe 0/1
    const int cb  = wave & 3;             // col block 0..3 (256 cols each)
    const int mr  = rt2 * 16 + l16;

    bf16x16 a0, a1;
#pragma unroll
    for (int e = 0; e < 8; ++e) {
        const int c0 =      16 * h + e;
        const int c1 =  8 + 16 * h + e;
        const int c2 = 32 + 16 * h + e;
        const int c3 = 40 + 16 * h + e;
        float s0 = vp[0][mr][c0] + vp[1][mr][c0] + vp[2][mr][c0] + vp[3][mr][c0] + bv[c0];
        float s1 = vp[0][mr][c1] + vp[1][mr][c1] + vp[2][mr][c1] + vp[3][mr][c1] + bv[c1];
        float s2 = vp[0][mr][c2] + vp[1][mr][c2] + vp[2][mr][c2] + vp[3][mr][c2] + bv[c2];
        float s3 = vp[0][mr][c3] + vp[1][mr][c3] + vp[2][mr][c3] + vp[3][mr][c3] + bv[c3];
        a0[e]     = (__bf16)s0;
        a0[8 + e] = (__bf16)s1;
        a1[e]     = (__bf16)s2;
        a1[8 + e] = (__bf16)s3;
    }

    const long orowbase = blockrow + rt2 * 16 + 8 * h;
    for (int nt = cb * 16; nt < cb * 16 + 16; ++nt) {
        const bf16x16 b0 = *(const bf16x16*)(WoT + (nt * 16 + l16) * HDIM + 16 * h);
        const bf16x16 b1 = *(const bf16x16*)(WoT + (nt * 16 + l16) * HDIM + 32 + 16 * h);
        f32x8 c = {};
        c = __builtin_amdgcn_wmma_f32_16x16x32_bf16(
                false, a0, false, b0, (short)0, c, false, false);
        c = __builtin_amdgcn_wmma_f32_16x16x32_bf16(
                false, a1, false, b1, (short)0, c, false, false);

        const int   col = nt * 16 + l16;
        const float boc = bo[col];
#pragma unroll
        for (int r = 0; r < 8; ++r)
            out[(orowbase + r) * DMODEL + col] = c[r] + boc;
    }
}

// ---------------------------------------------------------------------------
extern "C" void kernel_launch(void* const* d_in, const int* in_sizes, int n_in,
                              void* d_out, int out_size, void* d_ws, size_t ws_size,
                              hipStream_t stream) {
    // setup_inputs order: x, Wq, bq, Wk, bk, Wv, bv, Wo, bo
    const float* x  = (const float*)d_in[0];
    const float* Wv = (const float*)d_in[5];
    const float* bv = (const float*)d_in[6];
    const float* Wo = (const float*)d_in[7];
    const float* bo = (const float*)d_in[8];

    __bf16* WvT = (__bf16*)d_ws;                                   // 128 KB
    __bf16* WoT = (__bf16*)((char*)d_ws + HDIM * DMODEL * 2);      // 128 KB

    prep_weights<<<(2 * HDIM * DMODEL + 255) / 256, 256, 0, stream>>>(Wv, Wo, WvT, WoT);
    mqa_fused<<<NROWS / ROWS_PER_WG, 256, 0, stream>>>(x, WvT, bv, WoT, bo, (float*)d_out);
}
